// LSTMAutoEncoderWithAttention_63393717288978
// MI455X (gfx1250) — matrix operand order repaired and checked by reading, not compiled, not run on hardware
//
#include <hip/hip_runtime.h>
#include <cmath>

typedef unsigned short u16;
typedef unsigned int   u32;
typedef __attribute__((ext_vector_type(16))) __bf16 v16bf;
typedef __attribute__((ext_vector_type(8)))  float  v8f;
typedef __attribute__((ext_vector_type(8)))  u32    v8u;

#define TT  512
#define BB  256
#define DD  256
#define HH  256
#define G4H 1024
#define BH  (BB * HH)

// ---------------- bf16 split helpers ----------------
static __device__ __forceinline__ u16 bf16_rn(float x) {
  u32 u = __float_as_uint(x);
  u32 r = u + 0x7fffu + ((u >> 16) & 1u);
  return (u16)(r >> 16);
}
static __device__ __forceinline__ float bf16_to_f(u16 h) {
  return __uint_as_float(((u32)h) << 16);
}
static __device__ __forceinline__ void split2(float x, u16 &hi, u16 &lo) {
  u16 h = bf16_rn(x);
  hi = h;
  lo = bf16_rn(x - bf16_to_f(h));
}

// ---------------- hardware tanh / sigmoid (CDNA5 V_TANH_F32) ----------------
static __device__ __forceinline__ float htanh(float x) {
#if defined(__has_builtin) && __has_builtin(__builtin_amdgcn_tanhf)
  return __builtin_amdgcn_tanhf(x);
#else
  float r;
  // TRANS op: ISA requires 1 independent op / V_NOP before result use.
  asm volatile("v_tanh_f32 %0, %1\n\tv_nop\n\tv_nop" : "=v"(r) : "v"(x));
  return r;
#endif
}
static __device__ __forceinline__ float hsig(float x) {
  return fmaf(htanh(x * 0.5f), 0.5f, 0.5f); // sigmoid(x) = 0.5*(1+tanh(x/2))
}

// ---------------- conversion / prep kernels ----------------
__global__ __launch_bounds__(256) void convert_split(const float *__restrict__ src,
                                                     u16 *__restrict__ hi,
                                                     u16 *__restrict__ lo, size_t n) {
  size_t i = (size_t)blockIdx.x * blockDim.x + threadIdx.x;
  size_t stride = (size_t)gridDim.x * blockDim.x;
  for (; i < n; i += stride) {
    u16 h, l;
    split2(src[i], h, l);
    hi[i] = h;
    lo[i] = l;
  }
}

// W: [Nw, Kw] row-major weights. Output: fragment-major swizzled packed-pair
// layout so a lane's 8 B-fragment dwords are contiguous (2x b128 loads):
//   packed row pk = k/2 (0..255), decompose pk = kc*16 + hl*8 + v
//   dst_u32[ ((kc*2 + hl)*Ntot + n)*8 + v ] = {bf16(W[n][k]), bf16(W[n][k+1])}
__global__ __launch_bounds__(256) void convert_pack(const float *__restrict__ W,
                                                    u32 *__restrict__ hiP,
                                                    u32 *__restrict__ loP,
                                                    int Nw, int Kw, int Ntot, int kOff) {
  int idx = blockIdx.x * blockDim.x + threadIdx.x;
  int total = Nw * (Kw >> 1);
  if (idx >= total) return;
  int n = idx % Nw;
  int kp = idx / Nw;
  int k = kp * 2;
  u16 h0, l0, h1, l1;
  split2(W[(size_t)n * Kw + k], h0, l0);
  split2(W[(size_t)n * Kw + k + 1], h1, l1);
  int pk = kOff / 2 + kp; // global packed row, 0..255
  int kc = pk >> 4, r = pk & 15, hl2 = r >> 3, v = r & 7;
  size_t d = ((size_t)(kc * 2 + hl2) * Ntot + n) * 8 + v;
  hiP[d] = (u32)h0 | ((u32)h1 << 16);
  loP[d] = (u32)l0 | ((u32)l1 << 16);
}

// W: [Nw, Kw] -> transposed split [Kw, Nw] (plain row-major for GEMM B operand)
__global__ __launch_bounds__(256) void convert_split_T(const float *__restrict__ W,
                                                       u16 *__restrict__ hiT,
                                                       u16 *__restrict__ loT,
                                                       int Nw, int Kw) {
  int idx = blockIdx.x * blockDim.x + threadIdx.x;
  if (idx >= Nw * Kw) return;
  int n = idx % Nw;
  int k = idx / Nw;
  u16 h, l;
  split2(W[(size_t)n * Kw + k], h, l);
  hiT[(size_t)k * Nw + n] = h;
  loT[(size_t)k * Nw + n] = l;
}

__global__ __launch_bounds__(256) void prep_layer(const float *__restrict__ h0,
                                                  const float *__restrict__ c0,
                                                  const float *__restrict__ bih,
                                                  const float *__restrict__ bhh,
                                                  u16 *__restrict__ hs_hi,
                                                  u16 *__restrict__ hs_lo,
                                                  float *__restrict__ c,
                                                  float *__restrict__ bias) {
  int idx = blockIdx.x * blockDim.x + threadIdx.x;
  if (idx < BH) {
    u16 h, l;
    split2(h0[idx], h, l);
    hs_hi[idx] = h;
    hs_lo[idx] = l;
    c[idx] = c0[idx];
  }
  if (idx < G4H) bias[idx] = bih[idx] + bhh[idx];
}

// ---------------- generic split-3 bf16 WMMA GEMM ----------------
// C[M,N] = A[M,K] * B[K,N] ( + bias[n] ), A/B given as bf16 hi/lo pairs.
// OUTMODE 0: fp32 out (+bias);  OUTMODE 1: bf16 hi/lo split out.
template <int OUTMODE>
__global__ __launch_bounds__(128) void gemm_split3(
    const u16 *__restrict__ Ah, const u16 *__restrict__ Al,
    const u16 *__restrict__ Bh, const u16 *__restrict__ Bl,
    const float *__restrict__ bias, float *__restrict__ Cf,
    u16 *__restrict__ Chi, u16 *__restrict__ Clo, int M, int N, int K) {
  __shared__ __align__(16) u16 As_h[64][36], As_l[64][36]; // [m][k]
  __shared__ __align__(16) u16 Bs_h[64][36], Bs_l[64][36]; // [n][k] (transposed)

  const int tid = threadIdx.x;
  const int wave = tid >> 5, lane = tid & 31;
  const int n16 = lane & 15, hl = lane >> 4;
  const size_t tile_m = (size_t)blockIdx.x * 64;
  const size_t tile_n = (size_t)blockIdx.y * 64;

  v8f acc[4];
#pragma unroll
  for (int i = 0; i < 4; ++i)
#pragma unroll
    for (int j = 0; j < 8; ++j) acc[i][j] = 0.0f;

  const int koffA[8] = {0, 2, 4, 6, 16, 18, 20, 22};

  for (int kc = 0; kc < K; kc += 32) {
    // stage A tile (64 x 32), row-major in LDS
#pragma unroll
    for (int it = 0; it < 4; ++it) {
      int idx = tid + it * 128; // 512 quads of 4 u16
      int r = idx >> 3, q = (idx & 7) * 4;
      size_t g = (tile_m + r) * (size_t)K + kc + q;
      *(uint2 *)&As_h[r][q] = *(const uint2 *)(Ah + g);
      *(uint2 *)&As_l[r][q] = *(const uint2 *)(Al + g);
    }
    // stage B tile (32 x 64) transposed -> [n][k]
#pragma unroll
    for (int it = 0; it < 4; ++it) {
      int idx = tid + it * 128; // 512 quads
      int k = idx >> 4, nq = (idx & 15) * 4;
      size_t g = (size_t)(kc + k) * N + tile_n + nq;
      uint2 vh = *(const uint2 *)(Bh + g);
      uint2 vl = *(const uint2 *)(Bl + g);
      u16 e[4];
      *(uint2 *)e = vh;
      Bs_h[nq + 0][k] = e[0]; Bs_h[nq + 1][k] = e[1];
      Bs_h[nq + 2][k] = e[2]; Bs_h[nq + 3][k] = e[3];
      *(uint2 *)e = vl;
      Bs_l[nq + 0][k] = e[0]; Bs_l[nq + 1][k] = e[1];
      Bs_l[nq + 2][k] = e[2]; Bs_l[nq + 3][k] = e[3];
    }
    __syncthreads();

    v8u ahi, alo;
#pragma unroll
    for (int v = 0; v < 8; ++v) {
      int k = 2 * v + 16 * hl;
      ahi[v] = *(const u32 *)&As_h[wave * 16 + n16][k];
      alo[v] = *(const u32 *)&As_l[wave * 16 + n16][k];
    }
    v16bf a_h = __builtin_bit_cast(v16bf, ahi);
    v16bf a_l = __builtin_bit_cast(v16bf, alo);
#pragma unroll
    for (int ni = 0; ni < 4; ++ni) {
      v8u bhi, blo;
#pragma unroll
      for (int v = 0; v < 8; ++v) {
        int k = 2 * v + 16 * hl;
        bhi[v] = *(const u32 *)&Bs_h[ni * 16 + n16][k];
        blo[v] = *(const u32 *)&Bs_l[ni * 16 + n16][k];
      }
      v16bf b_h = __builtin_bit_cast(v16bf, bhi);
      v16bf b_l = __builtin_bit_cast(v16bf, blo);
      acc[ni] = __builtin_amdgcn_wmma_f32_16x16x32_bf16(false, a_h, false, b_h,
                                                        (short)0, acc[ni], false, false);
      acc[ni] = __builtin_amdgcn_wmma_f32_16x16x32_bf16(false, a_h, false, b_l,
                                                        (short)0, acc[ni], false, false);
      acc[ni] = __builtin_amdgcn_wmma_f32_16x16x32_bf16(false, a_l, false, b_h,
                                                        (short)0, acc[ni], false, false);
    }
    __syncthreads();
  }

#pragma unroll
  for (int ni = 0; ni < 4; ++ni) {
#pragma unroll
    for (int v = 0; v < 8; ++v) {
      size_t gm = tile_m + wave * 16 + v + 8 * hl;
      size_t gn = tile_n + ni * 16 + n16;
      float val = acc[ni][v];
      if (OUTMODE == 0) {
        if (bias) val += bias[gn];
        Cf[gm * N + gn] = val;
      } else {
        u16 h, l;
        split2(val, h, l);
        Chi[gm * N + gn] = h;
        Clo[gm * N + gn] = l;
      }
    }
  }
}

// ---------------- persistent fused LSTM kernel ----------------
// grid = 256 blocks x 32 threads. Block (rt,ct) owns batch rows rt*16..+15 and
// hidden cols ct*16..+15; computes all four gates for that tile each step:
//   gates = bias + [x_t , h_{t-1}] @ Wcat^T        (K = 512, split-3 bf16 WMMA)
// then applies the LSTM pointwise update (hardware V_TANH) and device-wide syncs.
__global__ __launch_bounds__(32) void lstm_persistent(
    const u16 *__restrict__ Xh, const u16 *__restrict__ Xl,  // [T,B,256] bf16 hi/lo
    const u32 *__restrict__ Wh, const u32 *__restrict__ Wlo, // swizzled packed weights
    const float *__restrict__ bias,                          // [1024]
    u16 *__restrict__ HsH, u16 *__restrict__ HsL,            // [(T+1),B,H]
    float *__restrict__ C,                                   // [B,H]
    u32 *sync_cnt, int Tsteps) {
  const int lane = threadIdx.x;
  const int n16 = lane & 15, hl = lane >> 4;
  const int rt = blockIdx.x & 15, ct = blockIdx.x >> 4;
  const int row0 = rt * 16, col0 = ct * 16;
  const int rowA = row0 + n16; // A-fragment row (m = lane%16)
  const int koffA[8] = {0, 2, 4, 6, 16, 18, 20, 22};

  for (int t = 0; t < Tsteps; ++t) {
    v8f acc[4];
#pragma unroll
    for (int g = 0; g < 4; ++g) {
      float bv = bias[g * HH + col0 + n16];
#pragma unroll
      for (int v = 0; v < 8; ++v) acc[g][v] = bv;
    }
    const u16 *xrow_h = Xh + (size_t)t * BB * DD + (size_t)rowA * DD;
    const u16 *xrow_l = Xl + (size_t)t * BB * DD + (size_t)rowA * DD;
    const u16 *hrow_h = HsH + (size_t)t * BH + (size_t)rowA * HH;
    const u16 *hrow_l = HsL + (size_t)t * BH + (size_t)rowA * HH;

    // prefetch next timestep's x rows (global_prefetch_b8) to hide HBM latency
    if (t + 1 < Tsteps) {
      __builtin_prefetch(xrow_h + (size_t)BB * DD, 0, 0);
      __builtin_prefetch(xrow_l + (size_t)BB * DD, 0, 0);
    }

    for (int kc = 0; kc < 16; ++kc) { // K = 512 in chunks of 32
      const u16 *ap_h = (kc < 8) ? xrow_h : hrow_h;
      const u16 *ap_l = (kc < 8) ? xrow_l : hrow_l;
      int kbase = (kc & 7) * 32;
      v8u ahi, alo;
#pragma unroll
      for (int v = 0; v < 8; ++v) {
        int k = kbase + 2 * v + 16 * hl;
        ahi[v] = *(const u32 *)(ap_h + k);
        alo[v] = *(const u32 *)(ap_l + k);
      }
      v16bf a_h = __builtin_bit_cast(v16bf, ahi);
      v16bf a_l = __builtin_bit_cast(v16bf, alo);
#pragma unroll
      for (int g = 0; g < 4; ++g) {
        // swizzled layout: lane's 8 fragment dwords are contiguous -> 2x b128
        size_t fb = ((size_t)(kc * 2 + hl) * G4H + g * HH + col0 + n16) * 8;
        const uint4 *ph = (const uint4 *)(Wh + fb);
        const uint4 *pl = (const uint4 *)(Wlo + fb);
        uint4 h0 = ph[0], h1 = ph[1];
        uint4 l0 = pl[0], l1 = pl[1];
        v8u bhi, blo;
        bhi[0] = h0.x; bhi[1] = h0.y; bhi[2] = h0.z; bhi[3] = h0.w;
        bhi[4] = h1.x; bhi[5] = h1.y; bhi[6] = h1.z; bhi[7] = h1.w;
        blo[0] = l0.x; blo[1] = l0.y; blo[2] = l0.z; blo[3] = l0.w;
        blo[4] = l1.x; blo[5] = l1.y; blo[6] = l1.z; blo[7] = l1.w;
        v16bf b_h = __builtin_bit_cast(v16bf, bhi);
        v16bf b_l = __builtin_bit_cast(v16bf, blo);
        acc[g] = __builtin_amdgcn_wmma_f32_16x16x32_bf16(false, a_h, false, b_h,
                                                         (short)0, acc[g], false, false);
        acc[g] = __builtin_amdgcn_wmma_f32_16x16x32_bf16(false, a_h, false, b_l,
                                                         (short)0, acc[g], false, false);
        acc[g] = __builtin_amdgcn_wmma_f32_16x16x32_bf16(false, a_l, false, b_h,
                                                         (short)0, acc[g], false, false);
      }
    }

    // pointwise LSTM update (gate order i,f,g,o), hardware tanh, no divergence
    size_t outbase = (size_t)(t + 1) * BH;
#pragma unroll
    for (int v = 0; v < 8; ++v) {
      int bi = row0 + v + 8 * hl;
      int hj = col0 + n16;
      size_t cidx = (size_t)bi * HH + hj;
      float iv = acc[0][v], fv = acc[1][v], gv = acc[2][v], ov = acc[3][v];
      float cn = hsig(fv) * C[cidx] + hsig(iv) * htanh(gv);
      float hv = hsig(ov) * htanh(cn);
      C[cidx] = cn;
      u16 hh, llo;
      split2(hv, hh, llo);
      HsH[outbase + cidx] = hh;
      HsL[outbase + cidx] = llo;
    }

    // device-wide step barrier
    __builtin_amdgcn_fence(__ATOMIC_RELEASE, "agent");
    if (lane == 0) {
      __hip_atomic_fetch_add(sync_cnt, 1u, __ATOMIC_RELAXED, __HIP_MEMORY_SCOPE_AGENT);
      u32 target = (u32)(t + 1) * (u32)gridDim.x;
      while (__hip_atomic_load(sync_cnt, __ATOMIC_RELAXED, __HIP_MEMORY_SCOPE_AGENT) < target)
        __builtin_amdgcn_s_sleep(1);
    }
    __builtin_amdgcn_fence(__ATOMIC_ACQUIRE, "agent");
  }
}

// ---------------- host-side orchestration ----------------
extern "C" void kernel_launch(void *const *d_in, const int *in_sizes, int n_in,
                              void *d_out, int out_size, void *d_ws, size_t ws_size,
                              hipStream_t stream) {
  (void)in_sizes; (void)n_in; (void)out_size; (void)ws_size;
  const float *inputs = (const float *)d_in[0];
  const float *h1_0 = (const float *)d_in[1];
  const float *c1_0 = (const float *)d_in[2];
  const float *h2_0 = (const float *)d_in[3];
  const float *c2_0 = (const float *)d_in[4];
  const float *W_ih1 = (const float *)d_in[5];
  const float *W_hh1 = (const float *)d_in[6];
  const float *b_ih1 = (const float *)d_in[7];
  const float *b_hh1 = (const float *)d_in[8];
  const float *W_ih2 = (const float *)d_in[9];
  const float *W_hh2 = (const float *)d_in[10];
  const float *b_ih2 = (const float *)d_in[11];
  const float *b_hh2 = (const float *)d_in[12];
  const float *attW = (const float *)d_in[13];
  const float *Wl = (const float *)d_in[14];
  const float *bl = (const float *)d_in[15];
  float *out = (float *)d_out;

  char *ws = (char *)d_ws;
  size_t off = 0;
  auto take = [&](size_t bytes) -> char * {
    char *p = ws + off;
    off += (bytes + 255) & ~(size_t)255;
    return p;
  };

  u16 *x_hi = (u16 *)take((size_t)TT * BB * DD * 2);
  u16 *x_lo = (u16 *)take((size_t)TT * BB * DD * 2);
  u16 *attn_hi = (u16 *)take((size_t)TT * BB * HH * 2);
  u16 *attn_lo = (u16 *)take((size_t)TT * BB * HH * 2);
  u16 *hs1_hi = (u16 *)take((size_t)(TT + 1) * BH * 2);
  u16 *hs1_lo = (u16 *)take((size_t)(TT + 1) * BH * 2);
  u16 *hs2_hi = (u16 *)take((size_t)(TT + 1) * BH * 2);
  u16 *hs2_lo = (u16 *)take((size_t)(TT + 1) * BH * 2);
  u16 *aw_hi = (u16 *)take((size_t)TT * TT * 2);
  u16 *aw_lo = (u16 *)take((size_t)TT * TT * 2);
  u32 *W1hP = (u32 *)take((size_t)256 * G4H * 4);
  u32 *W1lP = (u32 *)take((size_t)256 * G4H * 4);
  u32 *W2hP = (u32 *)take((size_t)256 * G4H * 4);
  u32 *W2lP = (u32 *)take((size_t)256 * G4H * 4);
  u16 *WlT_hi = (u16 *)take((size_t)256 * 256 * 2);
  u16 *WlT_lo = (u16 *)take((size_t)256 * 256 * 2);
  float *bias1 = (float *)take((size_t)G4H * 4);
  float *bias2 = (float *)take((size_t)G4H * 4);
  float *c1 = (float *)take((size_t)BH * 4);
  float *c2 = (float *)take((size_t)BH * 4);
  u32 *sync0 = (u32 *)take(256);
  u32 *sync1 = (u32 *)take(256);

  hipMemsetAsync(sync0, 0, 256, stream);
  hipMemsetAsync(sync1, 0, 256, stream);

  convert_split<<<2048, 256, 0, stream>>>(inputs, x_hi, x_lo, (size_t)TT * BB * DD);
  convert_split<<<512, 256, 0, stream>>>(attW, aw_hi, aw_lo, (size_t)TT * TT);
  convert_pack<<<512, 256, 0, stream>>>(W_ih1, W1hP, W1lP, G4H, DD, G4H, 0);
  convert_pack<<<512, 256, 0, stream>>>(W_hh1, W1hP, W1lP, G4H, HH, G4H, DD);
  convert_pack<<<512, 256, 0, stream>>>(W_ih2, W2hP, W2lP, G4H, HH, G4H, 0);
  convert_pack<<<512, 256, 0, stream>>>(W_hh2, W2hP, W2lP, G4H, HH, G4H, HH);
  convert_split_T<<<256, 256, 0, stream>>>(Wl, WlT_hi, WlT_lo, 256, 256);
  prep_layer<<<256, 256, 0, stream>>>(h1_0, c1_0, b_ih1, b_hh1, hs1_hi, hs1_lo, c1, bias1);
  prep_layer<<<256, 256, 0, stream>>>(h2_0, c2_0, b_ih2, b_hh2, hs2_hi, hs2_lo, c2, bias2);

  // encoder LSTM (persistent, 512 steps)
  lstm_persistent<<<256, 32, 0, stream>>>(x_hi, x_lo, W1hP, W1lP, bias1, hs1_hi,
                                          hs1_lo, c1, sync0, TT);
  // attention: attn[T, B*H] = attW[T,T] @ hs1[T, B*H]
  gemm_split3<1><<<dim3(TT / 64, (BB * HH) / 64), 128, 0, stream>>>(
      aw_hi, aw_lo, hs1_hi + BH, hs1_lo + BH, nullptr, nullptr, attn_hi, attn_lo,
      TT, BB * HH, TT);
  // decoder LSTM
  lstm_persistent<<<256, 32, 0, stream>>>(attn_hi, attn_lo, W2hP, W2lP, bias2,
                                          hs2_hi, hs2_lo, c2, sync1, TT);
  // out[T*B, O] = hs2[T*B, H] @ Wl^T + bl
  gemm_split3<0><<<dim3((TT * BB) / 64, 256 / 64), 128, 0, stream>>>(
      hs2_hi + BH, hs2_lo + BH, WlT_hi, WlT_lo, bl, out, nullptr, nullptr,
      TT * BB, 256, 256);
}
